// HMformer_635655160333
// MI455X (gfx1250) — compile-verified, matrix operand order checked by reading, not run on hardware
//
#include <hip/hip_runtime.h>
#include <math.h>

// ---------------------------------------------------------------------------
// CDNA5 (gfx1250) HMformer forward. All GEMMs -> v_wmma_f32_16x16x32_bf16.
// Double-buffered LDS software pipeline: next tile's global loads are issued
// before the current tile's WMMAs, bf16 convert + ds_store_b128 after, one
// workgroup barrier per K-step. B is transposed in LDS so both operand
// fragments are contiguous 16B ds_load_b128 reads; 80B row pitch keeps the
// 16-lane fragment reads bank-conflict-free. All index math is 32-bit
// (every buffer here is < 2^31 elements).
// Requirement (met by all call sites): A has unit column stride (aCS == 1).
// ---------------------------------------------------------------------------

typedef __attribute__((ext_vector_type(8)))  __bf16 v8bf;
typedef __attribute__((ext_vector_type(16))) __bf16 v16bf;
typedef __attribute__((ext_vector_type(8)))  float  v8f;
typedef __attribute__((ext_vector_type(4)))  float  v4f;

#define TM 128
#define TN 64
#define TK 32
#define AP 40   // LDS row pitch (bf16 elems): 80B, multiple of 16B, 20 banks
#define BP 40

static __device__ __forceinline__ v16bf cat8(v8bf a, v8bf b) {
  return __builtin_shufflevector(a, b, 0, 1, 2, 3, 4, 5, 6, 7,
                                 8, 9, 10, 11, 12, 13, 14, 15);
}

struct GemmArgs {
  const float* A; long long aB1, aB2; int aRS;          // aCS == 1 required
  const float* B; long long bB1, bB2; int bRS, bCS;
  float*       C; long long cB1, cB2; int cRS;
  const float* bias;
  const float* Res; long long rB1, rB2; int rRS;
  int M, N, K, inner;
  float alpha; int act, accum;
};

// C = epilogue(alpha * A@B + bias [+Res]); optional exact GELU; optional +=.
__global__ __launch_bounds__(256) void gemm_bf16_wmma(GemmArgs g)
{
  __shared__ __attribute__((aligned(16))) __bf16 As [2][TM][AP];
  __shared__ __attribute__((aligned(16))) __bf16 BsT[2][TN][BP];  // [n][k]

  const int bz    = blockIdx.z;
  const int outer = bz / g.inner;
  const int inner = bz % g.inner;
  const float* A = g.A + (long long)outer * g.aB1 + (long long)inner * g.aB2;
  const float* B = g.B + (long long)outer * g.bB1 + (long long)inner * g.bB2;
  float*       C = g.C + (long long)outer * g.cB1 + (long long)inner * g.cB2;
  const float* R = g.Res ? (g.Res + (long long)outer * g.rB1 + (long long)inner * g.rB2)
                         : (const float*)nullptr;

  const int m0   = blockIdx.y * TM;
  const int n0   = blockIdx.x * TN;
  const int tid  = threadIdx.x;
  const int lane = tid & 31;
  const int wave = tid >> 5;
  const int wm   = wave & 3;           // 4 waves along M: 32 rows each
  const int wn   = wave >> 2;          // 2 waves along N: 32 cols each
  const int fm    = lane & 15;
  const int khalf = (lane >> 4) * 8;   // CDNA5 16-bit operand K-half select

  // A staging map: thread -> (row, 16 consecutive k); 16B-aligned v4f loads
  const int sa_r = tid >> 1;
  const int sa_k = (tid & 1) * 16;
  // B staging map: thread -> (col n, 8 consecutive k); coalesced over n
  const int sb_n = tid & 63;
  const int sb_k = (tid >> 6) * 8;

  float fA[16];
  float fB[8];

  // Issue next tile's global loads into registers (32-bit offset math).
  auto stage_load = [&](int k0) {
    {
      const int gm = m0 + sa_r;
      if (gm < g.M && (k0 + sa_k) < g.K) {
        const v4f* p = (const v4f*)(A + gm * g.aRS + (k0 + sa_k));
        const v4f q0 = p[0], q1 = p[1], q2 = p[2], q3 = p[3];
        #pragma unroll
        for (int j = 0; j < 4; ++j) {
          fA[j] = q0[j]; fA[4 + j] = q1[j]; fA[8 + j] = q2[j]; fA[12 + j] = q3[j];
        }
      } else {
        #pragma unroll
        for (int j = 0; j < 16; ++j) fA[j] = 0.0f;
      }
    }
    {
      const int gn = n0 + sb_n;
      if (gn < g.N && (k0 + sb_k) < g.K) {
        if (g.bRS == 1) {            // contiguous k (e.g. K^T in scores GEMM)
          const v4f* p = (const v4f*)(B + (k0 + sb_k) + gn * g.bCS);
          const v4f q0 = p[0], q1 = p[1];
          #pragma unroll
          for (int j = 0; j < 4; ++j) { fB[j] = q0[j]; fB[4 + j] = q1[j]; }
        } else {                     // strided k; coalesced across the n-group
          const float* p = B + gn * g.bCS + (k0 + sb_k) * g.bRS;
          #pragma unroll
          for (int j = 0; j < 8; ++j) fB[j] = p[j * g.bRS];
        }
      } else {
        #pragma unroll
        for (int j = 0; j < 8; ++j) fB[j] = 0.0f;
      }
    }
  };
  // Convert fp32 -> bf16 and commit the staged tile to one LDS buffer.
  auto stage_store = [&](int buf) {
    v8bf a0, a1, b0;
    #pragma unroll
    for (int j = 0; j < 8; ++j) {
      a0[j] = (__bf16)fA[j];
      a1[j] = (__bf16)fA[8 + j];
      b0[j] = (__bf16)fB[j];
    }
    *(v8bf*)&As [buf][sa_r][sa_k]     = a0;
    *(v8bf*)&As [buf][sa_r][sa_k + 8] = a1;
    *(v8bf*)&BsT[buf][sb_n][sb_k]     = b0;
  };

  v8f acc[2][2] = {};

  // Prologue: fill buffer 0.
  stage_load(0);
  stage_store(0);
  __syncthreads();

  int cur = 0;
  for (int k0 = 0; k0 < g.K; k0 += TK) {
    const bool more = (k0 + TK) < g.K;
    if (more) stage_load(k0 + TK);   // global loads in flight during WMMAs

    // Fragments: contiguous 16B runs -> ds_load_b128.
    // lanes 0-15: K = {0..7, 16..23}; lanes 16-31: K = {8..15, 24..31}.
    v16bf af[2], bg[2];
    #pragma unroll
    for (int mt = 0; mt < 2; ++mt) {
      const int ar = wm * 32 + mt * 16 + fm;
      af[mt] = cat8(*(const v8bf*)&As[cur][ar][khalf],
                    *(const v8bf*)&As[cur][ar][16 + khalf]);
    }
    #pragma unroll
    for (int nt = 0; nt < 2; ++nt) {
      const int br = wn * 32 + nt * 16 + fm;
      bg[nt] = cat8(*(const v8bf*)&BsT[cur][br][khalf],
                    *(const v8bf*)&BsT[cur][br][16 + khalf]);
    }
    #pragma unroll
    for (int mt = 0; mt < 2; ++mt)
      #pragma unroll
      for (int nt = 0; nt < 2; ++nt)
        acc[mt][nt] = __builtin_amdgcn_wmma_f32_16x16x32_bf16(
            false, af[mt], false, bg[nt], (short)0, acc[mt][nt], false, false);

    if (more) stage_store(cur ^ 1);
    __syncthreads();
    cur ^= 1;
  }

  // Epilogue: C/D layout lanes 0-15 -> M=r, lanes 16-31 -> M=8+r.
  #pragma unroll
  for (int mt = 0; mt < 2; ++mt) {
    const int mrow = m0 + wm * 32 + mt * 16 + khalf;
    #pragma unroll
    for (int nt = 0; nt < 2; ++nt) {
      const int nn = n0 + wn * 32 + nt * 16 + fm;
      if (nn >= g.N) continue;
      const float bv = g.bias ? g.bias[nn] : 0.0f;
      const v8f ac = acc[mt][nt];
      #pragma unroll
      for (int r = 0; r < 8; ++r) {
        const int mm = mrow + r;
        if (mm >= g.M) continue;
        float v = ac[r] * g.alpha + bv;
        if (g.act == 1) v = 0.5f * v * (1.0f + erff(v * 0.70710678118654752f));
        if (R) v += R[mm * g.rRS + nn];
        float* cp = C + mm * g.cRS + nn;
        if (g.accum) *cp += v; else *cp = v;
      }
    }
  }
}

// Per-(b,m) instance normalization of the input series (B,L,M layout).
__global__ void norm_input_kernel(const float* __restrict__ x,
                                  float* __restrict__ xn,
                                  float* __restrict__ meanBM,
                                  float* __restrict__ stdBM)
{
  const int bm = blockIdx.x;           // 0..255
  const int b = bm >> 3, m = bm & 7;
  __shared__ float ss[256], ss2[256];
  __shared__ float smean, sinv;
  float s = 0.0f, s2 = 0.0f;
  for (int l = threadIdx.x; l < 512; l += 256) {
    const float v = x[(b * 512 + l) * 8 + m];
    s += v; s2 += v * v;
  }
  ss[threadIdx.x] = s; ss2[threadIdx.x] = s2;
  __syncthreads();
  for (int off = 128; off > 0; off >>= 1) {
    if (threadIdx.x < off) {
      ss[threadIdx.x]  += ss[threadIdx.x + off];
      ss2[threadIdx.x] += ss2[threadIdx.x + off];
    }
    __syncthreads();
  }
  if (threadIdx.x == 0) {
    const float mu  = ss[0] * (1.0f / 512.0f);
    const float var = ss2[0] * (1.0f / 512.0f) - mu * mu;
    const float sd  = sqrtf(var + 1e-5f);
    meanBM[bm] = mu; stdBM[bm] = sd;
    smean = mu; sinv = 1.0f / sd;
  }
  __syncthreads();
  for (int l = threadIdx.x; l < 512; l += 256)
    xn[bm * 512 + l] = (x[(b * 512 + l) * 8 + m] - smean) * sinv;
}

// Unfold into patches with edge-replication pad (clamp index to L-1).
__global__ void patch_kernel(const float* __restrict__ xn,
                             float* __restrict__ patches,
                             int n, int P, int st, long long total)
{
  const long long i = (long long)blockIdx.x * 256 + threadIdx.x;
  if (i >= total) return;
  const int j  = (int)(i % P);
  const long long t2 = i / P;
  const int t  = (int)(t2 % n);
  const int bm = (int)(t2 / n);
  int idx = t * st + j;
  if (idx > 511) idx = 511;
  patches[i] = xn[bm * 512 + idx];
}

// Row softmax; one thread per row (rows are short: n <= 64).
__global__ void softmax_kernel(float* __restrict__ S, long long rows, int n)
{
  const long long r = (long long)blockIdx.x * 256 + threadIdx.x;
  if (r >= rows) return;
  float* p = S + r * n;
  float mx = -3.4e38f;
  for (int j = 0; j < n; ++j) mx = fmaxf(mx, p[j]);
  float sum = 0.0f;
  for (int j = 0; j < n; ++j) { const float e = expf(p[j] - mx); p[j] = e; sum += e; }
  const float inv = 1.0f / sum;
  for (int j = 0; j < n; ++j) p[j] *= inv;
}

// BatchNorm batch statistics over all rows per channel.
__global__ void bn_stats_kernel(const float* __restrict__ X, int Rr, int C,
                                float* __restrict__ mean, float* __restrict__ rstd)
{
  const int c = blockIdx.x;
  __shared__ float ss[256], ss2[256];
  float s = 0.0f, s2 = 0.0f;
  for (int r = threadIdx.x; r < Rr; r += 256) {
    const float v = X[(long long)r * C + c];
    s += v; s2 += v * v;
  }
  ss[threadIdx.x] = s; ss2[threadIdx.x] = s2;
  __syncthreads();
  for (int off = 128; off > 0; off >>= 1) {
    if (threadIdx.x < off) {
      ss[threadIdx.x]  += ss[threadIdx.x + off];
      ss2[threadIdx.x] += ss2[threadIdx.x + off];
    }
    __syncthreads();
  }
  if (threadIdx.x == 0) {
    const float m = ss[0] / (float)Rr;
    const float v = ss2[0] / (float)Rr - m * m;
    mean[c] = m;
    rstd[c] = rsqrtf(v + 1e-5f);
  }
}

__global__ void bn_apply_kernel(float* __restrict__ X, long long total, int C,
                                const float* __restrict__ mean,
                                const float* __restrict__ rstd,
                                const float* __restrict__ g,
                                const float* __restrict__ b)
{
  const long long i = (long long)blockIdx.x * 256 + threadIdx.x;
  if (i >= total) return;
  const int c = (int)(i % C);
  X[i] = (X[i] - mean[c]) * rstd[c] * g[c] + b[c];
}

__global__ void zero_kernel(float* __restrict__ p, int nElems)
{
  const int i = blockIdx.x * 256 + threadIdx.x;
  if (i < nElems) p[i] = 0.0f;
}

// out[b, p, m] = acc[(b*8+m)*96 + p] * std[b,m] + mean[b,m]
__global__ void final_out_kernel(const float* __restrict__ acc,
                                 const float* __restrict__ meanBM,
                                 const float* __restrict__ stdBM,
                                 float* __restrict__ out)
{
  const int i = blockIdx.x * 256 + threadIdx.x;
  if (i >= 32 * 96 * 8) return;
  const int m = i & 7;
  const int p = (i >> 3) % 96;
  const int b = i / (96 * 8);
  const int bm = b * 8 + m;
  out[i] = acc[bm * 96 + p] * stdBM[bm] + meanBM[bm];
}

// ---------------------------------------------------------------------------

struct LayerP {
  const float *bk,*bo,*bq,*bv,*c1b,*c1w,*c2b,*c2w,*n1b,*n1g,*n2b,*n2g,*wk,*wo,*wq,*wv;
};
struct ScaleP {
  const float *conv_b,*conv_w,*emb_b,*emb_w,*norm_b,*norm_g,*proj_b,*proj_w;
  LayerP L[2];
};

static inline void launch_gemm(hipStream_t st,
  const float* A, long long aB1, long long aB2, int aRS,
  const float* B, long long bB1, long long bB2, int bRS, int bCS,
  float* C, long long cB1, long long cB2, int cRS,
  const float* bias,
  const float* Res, long long rB1, long long rB2, int rRS,
  int M, int N, int K, int batch, int inner,
  float alpha, int act, int accum)
{
  GemmArgs g = { A, aB1, aB2, aRS,
                 B, bB1, bB2, bRS, bCS,
                 C, cB1, cB2, cRS,
                 bias,
                 Res, rB1, rB2, rRS,
                 M, N, K, inner, alpha, act, accum };
  dim3 grid((unsigned)((N + TN - 1) / TN), (unsigned)((M + TM - 1) / TM), (unsigned)batch);
  gemm_bf16_wmma<<<grid, dim3(256, 1, 1), 0, st>>>(g);
}

static inline void bn_run(hipStream_t st, float* X, int Rr, int C,
                          const float* g, const float* b, float* m, float* r)
{
  bn_stats_kernel<<<dim3((unsigned)C), dim3(256), 0, st>>>(X, Rr, C, m, r);
  const long long tot = (long long)Rr * C;
  bn_apply_kernel<<<dim3((unsigned)((tot + 255) / 256)), dim3(256), 0, st>>>(X, tot, C, m, r, g, b);
}

extern "C" void kernel_launch(void* const* d_in, const int* in_sizes, int n_in,
                              void* d_out, int out_size, void* d_ws, size_t ws_size,
                              hipStream_t stream)
{
  (void)in_sizes; (void)n_in; (void)out_size; (void)ws_size;

  const float* x_enc = (const float*)d_in[0];
  int idx = 1;
  auto nxt = [&]() { return (const float*)d_in[idx++]; };

  // Params flattened in jax pytree order (dict keys sorted alphabetically).
  ScaleP sc[3];
  for (int i = 0; i < 3; ++i) {
    if (i < 2) { sc[i].conv_b = nxt(); sc[i].conv_w = nxt(); }
    else       { sc[i].conv_b = nullptr; sc[i].conv_w = nullptr; }
    sc[i].emb_b = nxt(); sc[i].emb_w = nxt();
    for (int l = 0; l < 2; ++l) {
      LayerP& q = sc[i].L[l];
      q.bk  = nxt(); q.bo  = nxt(); q.bq  = nxt(); q.bv  = nxt();
      q.c1b = nxt(); q.c1w = nxt(); q.c2b = nxt(); q.c2w = nxt();
      q.n1b = nxt(); q.n1g = nxt(); q.n2b = nxt(); q.n2g = nxt();
      q.wk  = nxt(); q.wo  = nxt(); q.wq  = nxt(); q.wv  = nxt();
    }
    sc[i].norm_b = nxt(); sc[i].norm_g = nxt();
    sc[i].proj_b = nxt(); sc[i].proj_w = nxt();
  }

  // Workspace carve (~153 MB of fp32).
  float* w = (float*)d_ws;
  size_t off = 0;
  auto carve = [&](size_t nElems) { float* p = w + off; off += (nElems + 63) & ~(size_t)63; return p; };
  float* meanBM  = carve(256);
  float* stdBM   = carve(256);
  float* bnMean  = carve(4096);
  float* bnRstd  = carve(4096);
  float* outacc  = carve(256 * 96);
  float* xnorm   = carve(256 * 512);
  float* patches = carve(256 * 1024);
  float* hbuf    = carve(4194304);   // 256 * n * D  (n*D == 16384 at all scales)
  float* qbuf    = carve(4194304);
  float* kbuf    = carve(4194304);
  float* vbuf    = carve(4194304);
  float* carry   = carve(4194304);
  float* midbuf  = carve(16777216);  // 256 * n * ff (n*ff == 65536 at all scales)
  float* scores  = midbuf;           // alias: lifetimes disjoint within a layer
  float* obuf    = qbuf;             // alias: q dead after softmax

  norm_input_kernel<<<dim3(256), dim3(256), 0, stream>>>(x_enc, xnorm, meanBM, stdBM);
  zero_kernel<<<dim3((256 * 96 + 255) / 256), dim3(256), 0, stream>>>(outacc, 256 * 96);

  for (int i = 0; i < 3; ++i) {
    const ScaleP& S = sc[i];
    const int s = 1 << i, D = 256 * s, FF = 1024 * s, P = 16 * s, st = 8 * s;
    const int n = 64 / s, Rr = 256 * n, E = D / 8;
    const long long nD = (long long)n * D;

    // Patch unfold.
    {
      const long long tot = 256LL * n * P;
      patch_kernel<<<dim3((unsigned)((tot + 255) / 256)), dim3(256), 0, stream>>>(
          xnorm, patches, n, P, st, tot);
    }
    // Value embedding (+ cross-scale carry residual for i>0).
    launch_gemm(stream, patches, 0, 0, P,
                        S.emb_w, 0, 0, D, 1,
                        hbuf, 0, 0, D, S.emb_b,
                        (i > 0) ? carry : nullptr, 0, 0, D,
                        Rr, D, P, 1, 1, 1.0f, 0, 0);

    for (int l = 0; l < 2; ++l) {
      const LayerP& Lp = S.L[l];
      // Q, K, V projections.
      launch_gemm(stream, hbuf, 0, 0, D, Lp.wq, 0, 0, D, 1, qbuf, 0, 0, D,
                  Lp.bq, nullptr, 0, 0, 0, Rr, D, D, 1, 1, 1.0f, 0, 0);
      launch_gemm(stream, hbuf, 0, 0, D, Lp.wk, 0, 0, D, 1, kbuf, 0, 0, D,
                  Lp.bk, nullptr, 0, 0, 0, Rr, D, D, 1, 1, 1.0f, 0, 0);
      launch_gemm(stream, hbuf, 0, 0, D, Lp.wv, 0, 0, D, 1, vbuf, 0, 0, D,
                  Lp.bv, nullptr, 0, 0, 0, Rr, D, D, 1, 1, 1.0f, 0, 0);
      // scores[b,h] = Q Kᵀ / sqrt(E)   (batch = 256 * 8 heads)
      launch_gemm(stream, qbuf, nD, E, D,
                          kbuf, nD, E, 1, D,
                          scores, 8LL * n * n, (long long)n * n, n,
                          nullptr, nullptr, 0, 0, 0,
                          n, n, E, 2048, 8, 1.0f / sqrtf((float)E), 0, 0);
      softmax_kernel<<<dim3((unsigned)((2048LL * n + 255) / 256)), dim3(256), 0, stream>>>(
          scores, 2048LL * n, n);
      // o[b,h] = A V
      launch_gemm(stream, scores, 8LL * n * n, (long long)n * n, n,
                          vbuf, nD, E, D, 1,
                          obuf, nD, E, D,
                          nullptr, nullptr, 0, 0, 0,
                          n, E, n, 2048, 8, 1.0f, 0, 0);
      // x = x + o @ Wo + bo  (residual fused in epilogue)
      launch_gemm(stream, obuf, 0, 0, D, Lp.wo, 0, 0, D, 1, hbuf, 0, 0, D,
                  Lp.bo, hbuf, 0, 0, D, Rr, D, D, 1, 1, 1.0f, 0, 0);
      bn_run(stream, hbuf, Rr, D, Lp.n1g, Lp.n1b, bnMean, bnRstd);
      // FFN: mid = GELU(x @ W1 + b1)
      launch_gemm(stream, hbuf, 0, 0, D, Lp.c1w, 0, 0, FF, 1, midbuf, 0, 0, FF,
                  Lp.c1b, nullptr, 0, 0, 0, Rr, FF, D, 1, 1, 1.0f, 1, 0);
      // x = x + mid @ W2 + b2
      launch_gemm(stream, midbuf, 0, 0, FF, Lp.c2w, 0, 0, D, 1, hbuf, 0, 0, D,
                  Lp.c2b, hbuf, 0, 0, D, Rr, D, FF, 1, 1, 1.0f, 0, 0);
      bn_run(stream, hbuf, Rr, D, Lp.n2g, Lp.n2b, bnMean, bnRstd);
    }

    // Encoder final BatchNorm.
    bn_run(stream, hbuf, Rr, D, S.norm_g, S.norm_b, bnMean, bnRstd);
    // Per-scale projection, accumulated across scales: outacc += h @ Wp + bp.
    launch_gemm(stream, hbuf, 0, 0, 16384, S.proj_w, 0, 0, 96, 1,
                outacc, 0, 0, 96, S.proj_b, nullptr, 0, 0, 0,
                256, 96, 16384, 1, 1, 1.0f, 0, 1);

    if (i < 2) {
      // Conv1d(D->2D, k=2, stride=2) as two strided GEMMs over the pair axis:
      // carry[r,o] = Σ_c h[r,2p+0,c] W(c,o,k=0) + Σ_c h[r,2p+1,c] W(c,o,k=1) + b.
      const int M2 = 256 * (n / 2), N2 = 2 * D;
      launch_gemm(stream, hbuf,     0, 0, 2 * D,
                          S.conv_w,     0, 0, 2, 2 * D,
                          carry, 0, 0, 2 * D, S.conv_b, nullptr, 0, 0, 0,
                          M2, N2, D, 1, 1, 1.0f, 0, 0);
      launch_gemm(stream, hbuf + D, 0, 0, 2 * D,
                          S.conv_w + 1, 0, 0, 2, 2 * D,
                          carry, 0, 0, 2 * D, nullptr, nullptr, 0, 0, 0,
                          M2, N2, D, 1, 1, 1.0f, 0, 1);
    }
  }

  final_out_kernel<<<dim3((32 * 96 * 8 + 255) / 256), dim3(256), 0, stream>>>(
      outacc, meanBM, stdBM, (float*)d_out);
}
